// FourierAdapter_18674517803071
// MI455X (gfx1250) — matrix operand order repaired and checked by reading, not compiled, not run on hardware
//
#include <hip/hip_runtime.h>
#include <cstdint>
#include <cstddef>

#define HIDDEN   1536
#define BOT      128
#define TM       64                  // rows per block (256 blocks -> fills the machine)
#define KC       32                  // K-chunk for down-proj
#define NKC      (HIDDEN / KC)       // 48
#define WU_ROWS  384                 // h rows staged per up-proj stage
#define NSTAGE   (HIDDEN / WU_ROWS)  // 4
#define LN_EPS   1e-5f

// region0 dword layout: buf k at k*BUFSTRIDE: A [64][16] dwords, then W [128][16] dwords
#define WOFF      1024               // dwords: A buffer size (64 rows * 16 dwords)
#define BUFSTRIDE 3072               // dwords: A(1024) + W(2048)

typedef __attribute__((ext_vector_type(16))) __bf16 v16bf;
typedef __attribute__((ext_vector_type(8)))  float  v8f;

union FragBF { uint32_t u[8]; uint4 q[2]; v16bf v; };

// two f32 -> packed bf16 dword using native converts (v_cvt_pk_bf16_f32)
__device__ __forceinline__ uint32_t cvt2_bf16(float lo, float hi) {
    union { __bf16 h[2]; uint32_t u; } r;
    r.h[0] = (__bf16)lo;
    r.h[1] = (__bf16)hi;
    return r.u;
}

// 16 f32 (4 float4) -> 8 packed dwords (2 uint4)
__device__ __forceinline__ void cvt16_bf16(uint4& q0, uint4& q1,
                                           float4 f0, float4 f1, float4 f2, float4 f3) {
    q0.x = cvt2_bf16(f0.x, f0.y); q0.y = cvt2_bf16(f0.z, f0.w);
    q0.z = cvt2_bf16(f1.x, f1.y); q0.w = cvt2_bf16(f1.z, f1.w);
    q1.x = cvt2_bf16(f2.x, f2.y); q1.y = cvt2_bf16(f2.z, f2.w);
    q1.z = cvt2_bf16(f3.x, f3.y); q1.w = cvt2_bf16(f3.z, f3.w);
}

// stage one down-proj K-chunk: waves 0-3 stage hidden rows, waves 4-7 stage W_down rows
__device__ __forceinline__ void stage_down(const float* __restrict__ hid,
                                           const float* __restrict__ wd,
                                           uint32_t* lds0, int bufOff,
                                           int rowBase, int t, int k0) {
    if (t < 128) {
        const int srow = t >> 1;            // 0..63
        const int sko  = (t & 1) * 16;      // float offset
        const int skq  = (t & 1) * 8;       // dword offset
        const float4* gA = (const float4*)(hid + (size_t)(rowBase + srow) * HIDDEN + k0 + sko);
        uint4 a0, a1;
        cvt16_bf16(a0, a1, gA[0], gA[1], gA[2], gA[3]);
        uint4* sA = (uint4*)(lds0 + bufOff + srow * 16 + skq);
        sA[0] = a0; sA[1] = a1;
    } else {
        const int n = t - 128;              // 0..127: full 32-float W_down row
        const float4* gW = (const float4*)(wd + (size_t)n * HIDDEN + k0);
        uint4 w0, w1, w2, w3;
        cvt16_bf16(w0, w1, gW[0], gW[1], gW[2], gW[3]);
        cvt16_bf16(w2, w3, gW[4], gW[5], gW[6], gW[7]);
        uint4* sW = (uint4*)(lds0 + bufOff + WOFF + n * 16);
        sW[0] = w0; sW[1] = w1; sW[2] = w2; sW[3] = w3;
    }
}

// stage Wu[h0s .. h0s+383][0..127] as packed bf16 rows (256B/row)
__device__ __forceinline__ void stage_wu(const float* __restrict__ wu,
                                         uint32_t* ldsWu, int h0s, int t) {
    const int dseg = t & 7;                 // 16-float segment of a Wu row
    const int rgrp = t >> 3;                // 0..31
    for (int it = 0; it < WU_ROWS / 32; ++it) {
        const int hrel = rgrp + it * 32;
        const float4* g = (const float4*)(wu + (size_t)(h0s + hrel) * BOT + dseg * 16);
        uint4 q0, q1;
        cvt16_bf16(q0, q1, g[0], g[1], g[2], g[3]);
        uint4* s = (uint4*)(ldsWu + hrel * 64 + dseg * 8);
        s[0] = q0; s[1] = q1;
    }
}

// ---------------------------------------------------------------------------
// Fully fused: out = LayerNorm(hidden + (hidden @ Wd^T) @ Wu^T) * gamma + beta
// (FFT->IFFT round trip in the reference is an exact identity.)
// 256 blocks x 256 threads (8 wave32). Wave pair shares 16 rows:
//   down-proj: pair splits the 8 bottleneck N-tiles (4 each)
//   up-proj  : pair splits h-tiles (alternating); two passes:
//     pass 1: accumulate per-row sum/sumsq (no store)
//     pass 2: recompute tiles, normalize, single write of d_out
// ---------------------------------------------------------------------------
__global__ __launch_bounds__(256)
void adapter_fused_kernel(const float* __restrict__ hid,
                          const float* __restrict__ wd,
                          const float* __restrict__ wu,
                          const float* __restrict__ gamma,
                          const float* __restrict__ beta,
                          float* __restrict__ out)
{
    // 128 KB static LDS -> 2 blocks per WGP (320 KB):
    //   [0, 32K)   : down-proj bf16 double-buffered staging (2 x 12KB),
    //                then X dump [64][128] f32, then gamma/beta (12KB) + row stats (1KB)
    //   [32K,128K) : Wu bf16 stage [384][128] bf16
    __shared__ __align__(16) unsigned char smem[131072];
    uint32_t* lds0  = (uint32_t*)smem;                    // region0 as dwords
    uint32_t* ldsWu = (uint32_t*)(smem + 32768);          // [384][64] dwords (bf16 pairs)
    float*    ldsX  = (float*)smem;                       // [64][128] f32 (reuse of region0)

    const int t    = threadIdx.x;
    const int wave = t >> 5;
    const int lane = t & 31;
    const int l16  = lane & 15;
    const int hih  = lane >> 4;          // 0: lanes 0-15, 1: lanes 16-31
    const int pairIdx = wave >> 1;       // 0..3: which 16-row group
    const int subN    = wave & 1;        // 0/1: which half of N-tiles / h-tiles
    const int rowBase = blockIdx.x * TM;

    v8f acc[4];
#pragma unroll
    for (int i = 0; i < 4; ++i) acc[i] = (v8f){0.f,0.f,0.f,0.f,0.f,0.f,0.f,0.f};

    // ---------------- down projection: X[m,d] = sum_h A[m,h] * Wd[d,h] ----------------
    stage_down(hid, wd, lds0, 0, rowBase, t, 0);     // prologue: chunk 0 -> buffer 0
    __syncthreads();

    for (int kc = 0; kc < NKC; ++kc) {
        const int curOff = (kc & 1) * BUFSTRIDE;
        if (kc + 1 < NKC) {  // stage next chunk (overlaps with WMMA below)
            stage_down(hid, wd, lds0, ((kc + 1) & 1) * BUFSTRIDE, rowBase, t, (kc + 1) * KC);
        }
        // A fragment (16x32 bf16): lane row m=l16; lo lanes K={0..7,16..23}, hi {8..15,24..31}
        FragBF af;
        {
            const uint32_t* p = lds0 + curOff + (pairIdx * 16 + l16) * 16 + hih * 4;
            af.q[0] = *(const uint4*)(p);
            af.q[1] = *(const uint4*)(p + 8);
        }
#pragma unroll
        for (int i = 0; i < 4; ++i) {
            // B fragment (32x16 bf16): lane col n; lo lanes K={0..7,16..23}, hi {8..15,24..31}
            FragBF bf;
            const int n = (subN * 4 + i) * 16 + l16;
            const uint32_t* p = lds0 + curOff + WOFF + n * 16 + hih * 4;
            bf.q[0] = *(const uint4*)(p);
            bf.q[1] = *(const uint4*)(p + 8);
            acc[i] = __builtin_amdgcn_wmma_f32_16x16x32_bf16(
                false, af.v, false, bf.v, (short)0, acc[i], false, false);
        }
        __syncthreads();
    }

    // ---------------- dump bottleneck X to LDS (C/D layout -> row-major f32) ----------------
#pragma unroll
    for (int i = 0; i < 4; ++i) {
#pragma unroll
        for (int r = 0; r < 8; ++r) {
            const int m = pairIdx * 16 + r + hih * 8;
            const int n = (subN * 4 + i) * 16 + l16;
            ldsX[m * BOT + n] = acc[i][r];
        }
    }
    __syncthreads();

    // ---------------- up-proj A fragments: X as bf16, 4 K-chunks of 32 (one-time pack) ----
    FragBF ua[4];
    {
        const int m = pairIdx * 16 + l16;
#pragma unroll
        for (int c = 0; c < 4; ++c) {
            const float* p = ldsX + m * BOT + c * 32 + (hih ? 8 : 0);
            float4 f0 = *(const float4*)(p);
            float4 f1 = *(const float4*)(p + 4);
            float4 f2 = *(const float4*)(p + 16);
            float4 f3 = *(const float4*)(p + 20);
            cvt16_bf16(ua[c].q[0], ua[c].q[1], f0, f1, f2, f3);
        }
    }
    __syncthreads();                       // region0 (X) now dead everywhere

    // stage gamma/beta into region0: gamma dwords [0,1536), beta [1536,3072), stats @ 3072
    {
        const float4* g4 = (const float4*)gamma;
        const float4* b4 = (const float4*)beta;
        float4* gl = (float4*)lds0;
        float4* bl = (float4*)(lds0 + 1536);
#pragma unroll
        for (int j = 0; j < 2; ++j) {
            const int i4 = j * 256 + t;
            if (i4 < HIDDEN / 4) { gl[i4] = g4[i4]; bl[i4] = b4[i4]; }
        }
    }

    const int rlbase = rowBase + pairIdx * 16 + hih * 8;

    // ---------------- pass 1: Y tiles -> per-row sum / sumsq (no store) ----------------
    float sAcc[8], qAcc[8];
#pragma unroll
    for (int r = 0; r < 8; ++r) { sAcc[r] = 0.f; qAcc[r] = 0.f; }

    for (int st = 0; st < NSTAGE; ++st) {
        const int h0s = st * WU_ROWS;
        __syncthreads();
        stage_wu(wu, ldsWu, h0s, t);
        __syncthreads();
        for (int j = 0; j < WU_ROWS / 32; ++j) {        // 12 h-tiles per wave (alternating)
            const int tt   = 2 * j + subN;
            const int hrel = tt * 16 + l16;
            v8f yacc = (v8f){0.f,0.f,0.f,0.f,0.f,0.f,0.f,0.f};
#pragma unroll
            for (int c = 0; c < 4; ++c) {
                FragBF bfr;
                const uint32_t* p = ldsWu + hrel * 64 + c * 16 + hih * 4;
                bfr.q[0] = *(const uint4*)(p);
                bfr.q[1] = *(const uint4*)(p + 8);
                yacc = __builtin_amdgcn_wmma_f32_16x16x32_bf16(
                    false, ua[c].v, false, bfr.v, (short)0, yacc, false, false);
            }
            const int hcol = h0s + tt * 16 + l16;
#pragma unroll
            for (int r = 0; r < 8; ++r) {
                const size_t idx = (size_t)(rlbase + r) * HIDDEN + hcol;
                const float y = yacc[r] + hid[idx];     // residual
                sAcc[r] += y;
                qAcc[r] += y * y;
            }
        }
    }

    // reduce across the 16 lanes of each half-wave (rows live per half-wave)
#pragma unroll
    for (int off = 1; off < 16; off <<= 1) {
#pragma unroll
        for (int r = 0; r < 8; ++r) {
            sAcc[r] += __shfl_xor(sAcc[r], off, 32);
            qAcc[r] += __shfl_xor(qAcc[r], off, 32);
        }
    }
    // combine the wave-pair halves (each covered alternating h-tiles) through LDS
    float* stats = (float*)smem + 3072;    // [2 kinds][2 subN][64 rows]
    if (l16 == 0) {
#pragma unroll
        for (int r = 0; r < 8; ++r) {
            const int row64 = pairIdx * 16 + r + hih * 8;
            stats[subN * 64 + row64]       = sAcc[r];
            stats[128 + subN * 64 + row64] = qAcc[r];
        }
    }
    __syncthreads();

    float mu8[8], rs8[8];
#pragma unroll
    for (int r = 0; r < 8; ++r) {
        const int row64 = pairIdx * 16 + r + hih * 8;
        const float s  = stats[row64] + stats[64 + row64];
        const float q  = stats[128 + row64] + stats[192 + row64];
        const float mu = s * (1.f / (float)HIDDEN);
        const float vr = q * (1.f / (float)HIDDEN) - mu * mu;
        mu8[r] = mu;
        rs8[r] = rsqrtf(vr + LN_EPS);
    }

    // ---------------- pass 2: recompute tiles, normalize, single store of out ----------
    const float* gLds = (const float*)smem;          // gamma
    const float* bLds = (const float*)smem + 1536;   // beta

    for (int st = 0; st < NSTAGE; ++st) {
        const int h0s = st * WU_ROWS;
        __syncthreads();
        stage_wu(wu, ldsWu, h0s, t);                 // restage (pass 1 consumed it)
        __syncthreads();
        for (int j = 0; j < WU_ROWS / 32; ++j) {
            const int tt   = 2 * j + subN;
            const int hrel = tt * 16 + l16;
            v8f yacc = (v8f){0.f,0.f,0.f,0.f,0.f,0.f,0.f,0.f};
#pragma unroll
            for (int c = 0; c < 4; ++c) {
                FragBF bfr;
                const uint32_t* p = ldsWu + hrel * 64 + c * 16 + hih * 4;
                bfr.q[0] = *(const uint4*)(p);
                bfr.q[1] = *(const uint4*)(p + 8);
                yacc = __builtin_amdgcn_wmma_f32_16x16x32_bf16(
                    false, ua[c].v, false, bfr.v, (short)0, yacc, false, false);
            }
            const int hcol = h0s + tt * 16 + l16;
            const float gv = gLds[hcol];             // LDS broadcast
            const float bv = bLds[hcol];
#pragma unroll
            for (int r = 0; r < 8; ++r) {
                const size_t idx = (size_t)(rlbase + r) * HIDDEN + hcol;
                const float y = yacc[r] + hid[idx];  // residual (L2-hot reload)
                out[idx] = (y - mu8[r]) * rs8[r] * gv + bv;
            }
        }
    }
}

// ---------------------------------------------------------------------------
extern "C" void kernel_launch(void* const* d_in, const int* in_sizes, int n_in,
                              void* d_out, int out_size, void* d_ws, size_t ws_size,
                              hipStream_t stream) {
    (void)in_sizes; (void)n_in; (void)out_size; (void)d_ws; (void)ws_size;
    const float* hid   = (const float*)d_in[0];   // [4,4096,1536]
    const float* wd    = (const float*)d_in[1];   // [128,1536]
    const float* wu    = (const float*)d_in[2];   // [1536,128]
    const float* gamma = (const float*)d_in[3];   // [1536]
    const float* beta  = (const float*)d_in[4];   // [1536]
    float* out = (float*)d_out;                   // [4,4096,1536]

    const int rows = 4 * 4096;                    // 16384
    adapter_fused_kernel<<<dim3(rows / TM), dim3(256), 0, stream>>>(hid, wd, wu, gamma, beta, out);
}
